// RowLSTM_35751307772246
// MI455X (gfx1250) — matrix operand order repaired and checked by reading, not compiled, not run on hardware
//
#include <hip/hip_runtime.h>

// ---------------------------------------------------------------------------
// RowLSTM for MI455X (gfx1250).
//  - All three GEMMs via v_wmma_f32_16x16x32_bf16.
//  - Single persistent kernel for the 64-row recurrence (device-wide counter
//    barrier, agent-scope acquire/release) instead of 64 launches.
//  - Per-step xs tile staged into LDS via the Tensor Data Mover when the
//    builtin is available (tensor_load_to_lds + s_wait_tensorcnt).
// ---------------------------------------------------------------------------

typedef __bf16 bf16;
typedef __attribute__((ext_vector_type(16))) __bf16 bf16x16;
typedef __attribute__((ext_vector_type(8)))  float   f32x8;
typedef __attribute__((ext_vector_type(4)))  unsigned int u32x4;
typedef __attribute__((ext_vector_type(4)))  int i32x4;
typedef __attribute__((ext_vector_type(8)))  int i32x8;

#define SDIM  256
#define CIND  256
#define BN    32
#define HD    64
#define WD    64
#define BW    2048      // BN*WD pixels per row
#define NG    1024      // 4*SDIM gate columns
#define PT    32        // pixels per workgroup in the recurrence
#define NBLK  64        // BW / PT persistent workgroups

#if defined(__has_builtin)
#  if __has_builtin(__builtin_amdgcn_tensor_load_to_lds)
#    define HAVE_TDM 1
#  endif
#endif
#ifndef HAVE_TDM
#  define HAVE_TDM 0
#endif

union Chunk { uint4 u; __bf16 h[8]; };

__device__ __forceinline__ bf16x16 make_frag(const Chunk& lo, const Chunk& hi) {
  bf16x16 v;
#pragma unroll
  for (int i = 0; i < 8; ++i) { v[i] = lo.h[i]; v[8 + i] = hi.h[i]; }
  return v;
}

__device__ __forceinline__ f32x8 wmma_bf16(bf16x16 a, bf16x16 b, f32x8 c) {
  return __builtin_amdgcn_wmma_f32_16x16x32_bf16(false, a, false, b, (short)0, c,
                                                 false, false);
}

// B fragment (32x16, KxN): lane holds col n0+lane%16, K = k0+8h+{0..7,16..23}.
// Weights are stored [N][K]-major (natural [out][in] layout), bf16.
__device__ __forceinline__ bf16x16 load_b(const bf16* __restrict__ w, int n0,
                                          int k0, int ldk, int lane) {
  int n  = n0 + (lane & 15);
  int kb = k0 + ((lane & 16) ? 8 : 0);
  const bf16* p = w + (size_t)n * ldk + kb;
  Chunk lo, hi;
  lo.u = *(const uint4*)(p);
  hi.u = *(const uint4*)(p + 16);
  return make_frag(lo, hi);
}

// A fragment from the bf16 hidden state with 3-tap im2col (K = tap*256 + si),
// zero-padded at the sequence ends (rows -1 and 2048).
__device__ __forceinline__ bf16x16 load_a_hconv(const bf16* __restrict__ h,
                                                int pix_m0, int k0, int lane) {
  int m   = pix_m0 + (lane & 15);
  int tap = k0 >> 8;                               // 0..2
  int si0 = (k0 & 255) + ((lane & 16) ? 8 : 0);
  int row = m + tap - 1;
  Chunk lo, hi;
  if (row >= 0 && row < BW) {
    const bf16* p = h + (size_t)row * SDIM + si0;
    lo.u = *(const uint4*)(p);
    hi.u = *(const uint4*)(p + 16);
  } else {
    lo.u.x = lo.u.y = lo.u.z = lo.u.w = 0u;
    hi.u = lo.u;
  }
  return make_frag(lo, hi);
}

// A fragment for the causal i2s conv: K<256 -> center tap x[w], K>=256 -> left
// tap x[w-1] (zero when w==0).  xTh is [pixel][ci] bf16 for one image row.
__device__ __forceinline__ bf16x16 load_a_i2s(const bf16* __restrict__ xTh,
                                              int pix_m0, int k0, int lane) {
  int m   = pix_m0 + (lane & 15);
  int tap = k0 >> 8;                               // 0 = center, 1 = left
  int si0 = (k0 & 255) + ((lane & 16) ? 8 : 0);
  int row = m - tap;
  bool valid = (tap == 0) || ((m & (WD - 1)) != 0);
  Chunk lo, hi;
  if (valid) {
    const bf16* p = xTh + (size_t)row * CIND + si0;
    lo.u = *(const uint4*)(p);
    hi.u = *(const uint4*)(p + 16);
  } else {
    lo.u.x = lo.u.y = lo.u.z = lo.u.w = 0u;
    hi.u = lo.u;
  }
  return make_frag(lo, hi);
}

// A fragment from an LDS tile (row-major [PT][256] bf16).
__device__ __forceinline__ bf16x16 load_a_lds(const bf16* uS, int m0, int k0,
                                              int lane) {
  int m  = m0 + (lane & 15);
  int kb = k0 + ((lane & 16) ? 8 : 0);
  const bf16* p = uS + m * SDIM + kb;
  Chunk lo, hi;
  lo.u = *(const uint4*)(p);
  hi.u = *(const uint4*)(p + 16);
  return make_frag(lo, hi);
}

__device__ __forceinline__ float sigf(float x) { return 1.f / (1.f + __expf(-x)); }
__device__ __forceinline__ float tanhfast(float x) {
  float e = __expf(2.f * x);
  return (e - 1.f) / (e + 1.f);
}

#if HAVE_TDM
// TDM descriptor for one contiguous 16KB (8192 x bf16) global->LDS tile.
// D# per ISA 8.3/8.4: group0 = {count=1, lds_addr, global_addr, type=2},
// group1 = {data_size=2B, tensor_dim0=tile_dim0=8192, dim1=1, stride=8192}.
__device__ __forceinline__ void tdm_load_16kb(unsigned lds_off,
                                              const void* src) {
  unsigned long long ga = (unsigned long long)(size_t)src;
  u32x4 g0;
  g0.x = 1u;                                            // count = 1
  g0.y = lds_off;                                       // lds_addr
  g0.z = (unsigned)(ga & 0xFFFFFFFFull);                // global_addr[31:0]
  g0.w = (unsigned)((ga >> 32) & 0x1FFFFFFull) | (2u << 30);  // [56:32]|type=2
  i32x8 g1;
  g1[0] = (int)(1u << 16);          // workgroup_mask=0, data_size=1 (2 bytes)
  g1[1] = (int)(0x2000u << 16);     // tensor_dim0[15:0]=8192 (bits 63:48)
  g1[2] = (int)(1u << 16);          // tensor_dim1=1        (bits 95:80)
  g1[3] = (int)(0x2000u << 16);     // tile_dim0=8192       (bits 127:112)
  g1[4] = 1;                        // tile_dim1=1          (bits 143:128)
  g1[5] = 8192;                     // tensor_dim0_stride   (bits 191:160)
  g1[6] = 0;
  g1[7] = 0;
  i32x4 z4 = {0, 0, 0, 0};
#if __has_include(<hip/amd_detail/amd_gfx1250_TDM.h>)
  i32x8 z8 = {0, 0, 0, 0, 0, 0, 0, 0};
  __builtin_amdgcn_tensor_load_to_lds(g0, g1, z4, z4, z8, 0);
#else
  __builtin_amdgcn_tensor_load_to_lds(g0, g1, z4, z4, 0);
#endif
}
#endif

// ---------------------------------------------------------------------------
// Prologue kernels
// ---------------------------------------------------------------------------

__global__ void prep_kernel(const float* __restrict__ w_i2s,
                            const float* __restrict__ w_s2s,
                            const float* __restrict__ w_ih,
                            const float* __restrict__ b_ih,
                            const float* __restrict__ b_hh,
                            bf16* __restrict__ Wi2s, bf16* __restrict__ Ws2sT,
                            bf16* __restrict__ Wih, float* __restrict__ bias_g) {
  int tid = blockIdx.x * blockDim.x + threadIdx.x;
  int nthr = gridDim.x * blockDim.x;
  for (int i = tid; i < SDIM * 512; i += nthr) {
    int s = i >> 9, k = i & 511;
    float v;
    if (k < 256) {  // center tap with causal channel mask
      int si = k;
      float m = ((s % 3) < (si % 3)) ? 0.f : 1.f;
      v = w_i2s[(s * CIND + si) * 3 + 1] * m;
    } else {        // left tap (tap 0); tap 2 masked to zero entirely
      int si = k - 256;
      v = w_i2s[(s * CIND + si) * 3 + 0];
    }
    Wi2s[i] = (bf16)v;
  }
  for (int i = tid; i < SDIM * 768; i += nthr) {
    int s = i / 768, k = i % 768;
    int t = k >> 8, si = k & 255;
    Ws2sT[i] = (bf16)w_s2s[(s * SDIM + si) * 3 + t];
  }
  for (int i = tid; i < NG * SDIM; i += nthr) Wih[i] = (bf16)w_ih[i];
  for (int i = tid; i < NG; i += nthr) bias_g[i] = b_ih[i] + b_hh[i];
}

// NCHW fp32 -> (h, pixel, ci) bf16, writes coalesced along ci.
__global__ void transpose_kernel(const float* __restrict__ x,
                                 bf16* __restrict__ xT) {
  size_t total = (size_t)HD * BN * WD * CIND;
  for (size_t i = (size_t)blockIdx.x * blockDim.x + threadIdx.x; i < total;
       i += (size_t)gridDim.x * blockDim.x) {
    int ci = (int)(i & 255);
    int w  = (int)((i >> 8) & 63);
    int b  = (int)((i >> 14) & 31);
    int h  = (int)(i >> 19);
    xT[i] = (bf16)x[(((size_t)b * CIND + ci) * HD + h) * WD + w];
  }
}

__global__ void init_kernel(bf16* __restrict__ h0, float* __restrict__ c,
                            unsigned* __restrict__ counter) {
  int i = blockIdx.x * blockDim.x + threadIdx.x;
  if (i < BW * SDIM) { h0[i] = (bf16)0.f; c[i] = 0.f; }
  if (i == 0) *counter = 0u;
}

// xs[h][p][s] = causal conv + b_i2s, as one big WMMA GEMM (K = 512).
__global__ void __launch_bounds__(256)
i2s_kernel(const bf16* __restrict__ xT, const bf16* __restrict__ Wi2s,
           const float* __restrict__ b_i2s, bf16* __restrict__ xs) {
  const int h    = blockIdx.y;
  const int p0   = blockIdx.x * 128;
  const int wave = threadIdx.x >> 5;
  const int lane = threadIdx.x & 31;
  const int m0   = p0 + wave * 16;
  const bf16* xTh = xT + (size_t)h * BW * CIND;
  for (int nt = 0; nt < 16; ++nt) {
    const int n0 = nt * 16;
    f32x8 acc = {};
#pragma unroll
    for (int kc = 0; kc < 16; ++kc) {
      const int k0 = kc * 32;
      bf16x16 a = load_a_i2s(xTh, m0, k0, lane);
      bf16x16 b = load_b(Wi2s, n0, k0, 512, lane);
      acc = wmma_bf16(a, b, acc);
    }
    const int col = n0 + (lane & 15);
    const float bias = b_i2s[col];
#pragma unroll
    for (int rr = 0; rr < 8; ++rr) {
      int row = m0 + rr + ((lane & 16) ? 8 : 0);
      xs[((size_t)h * BW + row) * SDIM + col] = (bf16)(acc[rr] + bias);
    }
  }
}

// ---------------------------------------------------------------------------
// Persistent recurrence kernel: 64 blocks x 256 threads, all 64 rows in one
// launch, device-wide counter barrier between rows.
//   phase 1: u = xs[r] + b_s2s + conv3(h_prev)  (K=768 WMMA GEMM -> LDS bf16)
//   phase 2: gates = u @ w_ih^T + bias (K=256, 4 gates share A fragments),
//            cell update in registers, write h/c state + fp32 NCHW output.
// ---------------------------------------------------------------------------
__global__ void __launch_bounds__(256)
lstm_kernel(const bf16* __restrict__ xs, bf16* __restrict__ hA,
            bf16* __restrict__ hB, float* __restrict__ c,
            const bf16* __restrict__ Ws2sT, const float* __restrict__ b_s2s,
            const bf16* __restrict__ Wih, const float* __restrict__ bias_g,
            float* __restrict__ out, unsigned* __restrict__ counter) {
  __shared__ bf16 uS[PT * SDIM];    // 16 KB u tile
  __shared__ bf16 xsS[PT * SDIM];   // 16 KB xs tile (TDM destination)
  const int p0   = blockIdx.x * PT;
  const int wave = threadIdx.x >> 5;
  const int lane = threadIdx.x & 31;
  const int m0   = (wave & 1) * 16;      // 16-row strip inside the 32-pixel tile
  const int ngrp = wave >> 1;            // 0..3: column group

  bf16* hp = hA;
  bf16* hn = hB;

  for (int r = 0; r < HD; ++r) {
    const bf16* xsr = xs + ((size_t)r * BW + p0) * SDIM;

    // Stage this block's contiguous 16KB xs tile into LDS.
#if HAVE_TDM
    if (threadIdx.x < 32) {
      tdm_load_16kb((unsigned)(size_t)&xsS[0], xsr);
      __builtin_amdgcn_s_wait_tensorcnt(0);
    }
#else
    for (int i = (int)threadIdx.x; i < PT * SDIM / 8; i += 256)
      ((uint4*)xsS)[i] = ((const uint4*)xsr)[i];
#endif
    __syncthreads();

    // ---- phase 1: hidden 3-tap conv -> u tile in LDS ----
    for (int i = 0; i < 4; ++i) {
      const int n0 = (ngrp * 4 + i) * 16;
      f32x8 acc = {};
#pragma unroll
      for (int kc = 0; kc < 24; ++kc) {
        const int k0 = kc * 32;
        bf16x16 a = load_a_hconv(hp, p0 + m0, k0, lane);
        bf16x16 b = load_b(Ws2sT, n0, k0, 768, lane);
        acc = wmma_bf16(a, b, acc);
      }
      const int col = n0 + (lane & 15);
      const float bias = b_s2s[col];
#pragma unroll
      for (int rr = 0; rr < 8; ++rr) {
        int lrow = m0 + rr + ((lane & 16) ? 8 : 0);
        float xv = (float)xsS[lrow * SDIM + col];
        uS[lrow * SDIM + col] = (bf16)(acc[rr] + xv + bias);
      }
    }
    __syncthreads();

    if (threadIdx.x == 0 && r + 1 < HD)  // warm L2 for the next row's tile
      __builtin_prefetch(xs + ((size_t)(r + 1) * BW + p0) * SDIM, 0, 1);

    // ---- phase 2: gate GEMM + cell update ----
    bf16x16 afr[8];
#pragma unroll
    for (int kc = 0; kc < 8; ++kc) afr[kc] = load_a_lds(uS, m0, kc * 32, lane);

    for (int i = 0; i < 4; ++i) {
      const int n0 = (ngrp * 4 + i) * 16;
      f32x8 ai = {}, af = {}, ag = {}, ao = {};
#pragma unroll
      for (int kc = 0; kc < 8; ++kc) {
        const int k0 = kc * 32;
        ai = wmma_bf16(afr[kc], load_b(Wih, 0 * SDIM + n0, k0, SDIM, lane), ai);
        af = wmma_bf16(afr[kc], load_b(Wih, 1 * SDIM + n0, k0, SDIM, lane), af);
        ag = wmma_bf16(afr[kc], load_b(Wih, 2 * SDIM + n0, k0, SDIM, lane), ag);
        ao = wmma_bf16(afr[kc], load_b(Wih, 3 * SDIM + n0, k0, SDIM, lane), ao);
      }
      const int col = n0 + (lane & 15);
      const float bi  = bias_g[0 * SDIM + col];
      const float bff = bias_g[1 * SDIM + col];
      const float bg  = bias_g[2 * SDIM + col];
      const float bo  = bias_g[3 * SDIM + col];
#pragma unroll
      for (int rr = 0; rr < 8; ++rr) {
        int lrow = m0 + rr + ((lane & 16) ? 8 : 0);
        int p = p0 + lrow;
        size_t ci = (size_t)p * SDIM + col;
        float iv = sigf(ai[rr] + bi);
        float fv = sigf(af[rr] + bff);
        float gv = tanhfast(ag[rr] + bg);
        float ov = sigf(ao[rr] + bo);
        float cn = fv * c[ci] + iv * gv;
        c[ci] = cn;
        float hn_ = ov * tanhfast(cn);
        hn[ci] = (bf16)hn_;
        int bb = p >> 6, ww = p & 63;
        out[(((size_t)bb * SDIM + col) * HD + r) * WD + ww] = hn_;
      }
    }

    // ---- device-wide barrier before the next row consumes hn ----
    __threadfence();                          // release this block's h writes
    __syncthreads();
    if (threadIdx.x == 0) {
      __hip_atomic_fetch_add(counter, 1u, __ATOMIC_RELEASE,
                             __HIP_MEMORY_SCOPE_AGENT);
      unsigned tgt = (unsigned)(r + 1) * gridDim.x;
      while (__hip_atomic_load(counter, __ATOMIC_ACQUIRE,
                               __HIP_MEMORY_SCOPE_AGENT) < tgt)
        __builtin_amdgcn_s_sleep(1);
    }
    __syncthreads();
    __threadfence();                          // acquire other blocks' h writes

    bf16* t = hp; hp = hn; hn = t;
  }
}

// ---------------------------------------------------------------------------
// Host launcher
// ---------------------------------------------------------------------------
extern "C" void kernel_launch(void* const* d_in, const int* in_sizes, int n_in,
                              void* d_out, int out_size, void* d_ws,
                              size_t ws_size, hipStream_t stream) {
  const float* x     = (const float*)d_in[0];
  const float* w_i2s = (const float*)d_in[1];
  const float* b_i2s = (const float*)d_in[2];
  const float* w_s2s = (const float*)d_in[3];
  const float* b_s2s = (const float*)d_in[4];
  const float* w_ih  = (const float*)d_in[5];
  const float* b_ih  = (const float*)d_in[6];
  const float* b_hh  = (const float*)d_in[7];
  float* out = (float*)d_out;

  char* ws = (char*)d_ws;
  bf16*  xT      = (bf16*)(ws);                    // 64*2048*256 bf16 = 64 MB
  bf16*  xs      = (bf16*)(ws + 67108864);         // 64 MB
  bf16*  Wi2s    = (bf16*)(ws + 134217728);        // 256*512 bf16
  bf16*  Ws2sT   = (bf16*)(ws + 134479872);        // 256*768 bf16
  bf16*  Wih     = (bf16*)(ws + 134873088);        // 1024*256 bf16
  float* bias_g  = (float*)(ws + 135397376);       // 1024 f32
  bf16*  hping   = (bf16*)(ws + 135401472);        // 2048*256 bf16
  bf16*  hpong   = (bf16*)(ws + 136450048);        // 2048*256 bf16
  float* cbuf    = (float*)(ws + 137498624);       // 2048*256 f32
  unsigned* cnt  = (unsigned*)(ws + 139595776);    // barrier counter

  prep_kernel<<<256, 256, 0, stream>>>(w_i2s, w_s2s, w_ih, b_ih, b_hh, Wi2s,
                                       Ws2sT, Wih, bias_g);
  transpose_kernel<<<8192, 256, 0, stream>>>(x, xT);
  init_kernel<<<2048, 256, 0, stream>>>(hping, cbuf, cnt);
  i2s_kernel<<<dim3(16, 64), 256, 0, stream>>>(xT, Wi2s, b_i2s, xs);
  lstm_kernel<<<NBLK, 256, 0, stream>>>(xs, hping, hpong, cbuf, Ws2sT, b_s2s,
                                        Wih, bias_g, out, cnt);
}
